// SeastarTGCN_55130200211791
// MI455X (gfx1250) — matrix operand order repaired and checked by reading, not compiled, not run on hardware
//
#include <hip/hip_runtime.h>
#include <math.h>

#define N_NODES 100000
#define N_EDGES 1600000
#define NT_TILES ((N_NODES + 15) / 16)   // 6250 exactly
#define WAVES_PER_BLOCK 8

typedef _Float16 half16 __attribute__((ext_vector_type(16)));
typedef float    floatx8 __attribute__((ext_vector_type(8)));

static __device__ __forceinline__ floatx8 wmma_f16(half16 a, half16 b, floatx8 c) {
  // D = A(16x32 f16) * B(32x16 f16) + C(16x16 f32)
  return __builtin_amdgcn_wmma_f32_16x16x32_f16(false, a, false, b, (short)0, c,
                                                false, false);
}

static __device__ __forceinline__ void lds_fence_wave() {
  // LDS ops are in-order per wave; wait for outstanding DS ops so cross-lane
  // staging through per-wave LDS is visible. "memory" stops compiler reordering.
  asm volatile("s_wait_dscnt 0" ::: "memory");
}

// Guaranteed hardware f32 atomic adds (no CAS-loop fallback risk).
// Store-class atomic (no return) -> STOREcnt; drained by s_endpgm's wait-idle.
static __device__ __forceinline__ void gatomic_add4_f32(float* p, float4 v) {
  asm volatile(
      "global_atomic_add_f32 %0, %1, off\n\t"
      "global_atomic_add_f32 %0, %2, off offset:4\n\t"
      "global_atomic_add_f32 %0, %3, off offset:8\n\t"
      "global_atomic_add_f32 %0, %4, off offset:12"
      :: "v"(p), "v"(v.x), "v"(v.y), "v"(v.z), "v"(v.w)
      : "memory");
}

// ---------------------------------------------------------------------------
// Kernel 1: H3[N,96] = f16(X[N,32]) @ f16([Wzc | Wrc | Whc])   (no bias yet)
// One wave per 16-row tile; 6 WMMAs (96/16 col tiles).
// ---------------------------------------------------------------------------
__global__ __launch_bounds__(256) void conv_pre_kernel(
    const float* __restrict__ X,
    const float* __restrict__ Wzc, const float* __restrict__ Wrc,
    const float* __restrict__ Whc,
    float* __restrict__ H3) {
  const int lane = threadIdx.x & 31;
  const int wave = threadIdx.x >> 5;
  int tile = blockIdx.x * WAVES_PER_BLOCK + wave;
  const bool active = (tile < NT_TILES);
  if (!active) tile = NT_TILES - 1;               // wave-uniform clamp
  const int base = tile * 16;
  const int m    = lane & 15;
  const int hi   = lane >> 4;                     // 0 | 1
  const int kb8  = hi * 8;
  const int kb16 = hi * 16;

  // A operand: 16x32 f16, ISA layout
  const float* xrow = X + (size_t)(base + m) * 32;
  half16 a;
#pragma unroll
  for (int j = 0; j < 8; ++j) {
    a[j]     = (_Float16)xrow[kb8 + j];
    a[8 + j] = (_Float16)xrow[16 + kb8 + j];
  }

  const float* Ws[3] = {Wzc, Wrc, Whc};
#pragma unroll
  for (int t = 0; t < 6; ++t) {
    const float* W = Ws[t >> 1];
    const int coff = (t & 1) * 16;
    half16 b;
#pragma unroll
    for (int i = 0; i < 16; ++i)
      b[i] = (_Float16)W[(kb8 + i + (i & 8)) * 32 + coff + m];
    floatx8 acc = {};
    acc = wmma_f16(a, b, acc);
    if (active) {
#pragma unroll
      for (int i = 0; i < 8; ++i) {
        int M = i + hi * 8;
        H3[(size_t)(base + M) * 96 + t * 16 + m] = acc[i];
      }
    }
  }
}

// ---------------------------------------------------------------------------
// Kernel 2: fused edge pass for all three GCNs.
// AGG[dst, 0:96] += ew * H3[src, 0:96].  One thread = (edge, 4-float chunk).
// H3 (38.4MB) and AGG (38.4MB) both fit in the 192MB L2 -> L2-resident.
// ---------------------------------------------------------------------------
__global__ __launch_bounds__(256) void edge_scatter_kernel(
    const float* __restrict__ H3, const float* __restrict__ ew,
    const int* __restrict__ src, const int* __restrict__ dst,
    float* __restrict__ AGG) {
  long long gid = (long long)blockIdx.x * blockDim.x + threadIdx.x;
  const long long total = (long long)N_EDGES * 24;   // 96 floats / 4 per thread
  if (gid >= total) return;
  int e = (int)(gid / 24);
  int c = (int)(gid % 24);
  int s = src[e], d = dst[e];
  float w = ew[e];
  float4 hv = *(const float4*)(H3 + (size_t)s * 96 + c * 4);
  hv.x *= w; hv.y *= w; hv.z *= w; hv.w *= w;
  gatomic_add4_f32(AGG + (size_t)d * 96 + c * 4, hv);
}

// ---------------------------------------------------------------------------
// Kernel 3: fused GRU gating + output head. One wave per 16-node tile.
// ---------------------------------------------------------------------------
__global__ __launch_bounds__(256) void gates_kernel(
    const float* __restrict__ AGG, const float* __restrict__ hidden,
    const float* __restrict__ bzc, const float* __restrict__ brc,
    const float* __restrict__ bhc,
    const float* __restrict__ Wz, const float* __restrict__ bz,
    const float* __restrict__ Wr, const float* __restrict__ br,
    const float* __restrict__ Wh, const float* __restrict__ bh,
    const float* __restrict__ Wlin, const float* __restrict__ blin,
    float* __restrict__ out_y, float* __restrict__ out_h) {
  __shared__ float stage[WAVES_PER_BLOCK * 16 * 32];
  const int lane = threadIdx.x & 31;
  const int wave = threadIdx.x >> 5;
  float* st = stage + wave * (16 * 32);           // private per-wave region
  int tile = blockIdx.x * WAVES_PER_BLOCK + wave;
  const bool active = (tile < NT_TILES);
  if (!active) tile = NT_TILES - 1;
  const int base = tile * 16;
  const int m    = lane & 15;
  const int hi   = lane >> 4;
  const int kb8  = hi * 8;
  const int kb16 = hi * 16;

  const size_t r96 = (size_t)(base + m) * 96;
  const size_t r32 = (size_t)(base + m) * 32;

  // A operands: hidden, cz/cr/ch (= AGG + conv bias) in f16 A-layout
  half16 a_h, a_cz, a_cr, a_ch;
#pragma unroll
  for (int j = 0; j < 8; ++j) {
    const int k0 = kb8 + j, k1 = 16 + kb8 + j;
    a_h[j]      = (_Float16)hidden[r32 + k0];
    a_h[8 + j]  = (_Float16)hidden[r32 + k1];
    a_cz[j]     = (_Float16)(AGG[r96 + k0] + bzc[k0]);
    a_cz[8 + j] = (_Float16)(AGG[r96 + k1] + bzc[k1]);
    a_cr[j]     = (_Float16)(AGG[r96 + 32 + k0] + brc[k0]);
    a_cr[8 + j] = (_Float16)(AGG[r96 + 32 + k1] + brc[k1]);
    a_ch[j]     = (_Float16)(AGG[r96 + 64 + k0] + bhc[k0]);
    a_ch[8 + j] = (_Float16)(AGG[r96 + 64 + k1] + bhc[k1]);
  }

  // B tile loader: W is [krows x 32] row-major; pick K-chunk base + 16-col tile
  auto loadB = [&](const float* W, int krow, int coff) -> half16 {
    half16 b;
#pragma unroll
    for (int i = 0; i < 16; ++i)
      b[i] = (_Float16)W[(krow + kb8 + i + (i & 8)) * 32 + coff + m];
    return b;
  };

  float zD[2][8], rD[2][8], hD[2][8];
#pragma unroll
  for (int nt = 0; nt < 2; ++nt) {
    const int coff = nt * 16;
    // z = sigmoid([cz | h] @ Wz + bz)
    floatx8 az = {};
    az = wmma_f16(a_cz, loadB(Wz, 0, coff), az);
    az = wmma_f16(a_h,  loadB(Wz, 32, coff), az);
    const float bvz = bz[coff + m];
#pragma unroll
    for (int i = 0; i < 8; ++i)
      zD[nt][i] = 1.0f / (1.0f + __expf(-(az[i] + bvz)));
    // r = sigmoid([cr | h] @ Wr + br); stage h*r into LDS
    floatx8 ar = {};
    ar = wmma_f16(a_cr, loadB(Wr, 0, coff), ar);
    ar = wmma_f16(a_h,  loadB(Wr, 32, coff), ar);
    const float bvr = br[coff + m];
#pragma unroll
    for (int i = 0; i < 8; ++i) {
      rD[nt][i] = 1.0f / (1.0f + __expf(-(ar[i] + bvr)));
      const int M = i + hi * 8;
      hD[nt][i] = hidden[(size_t)(base + M) * 32 + coff + m];
      st[M * 32 + coff + m] = hD[nt][i] * rD[nt][i];
    }
  }
  lds_fence_wave();

  half16 a_hr;
#pragma unroll
  for (int j = 0; j < 8; ++j) {
    a_hr[j]     = (_Float16)st[m * 32 + kb8 + j];
    a_hr[8 + j] = (_Float16)st[m * 32 + 16 + kb8 + j];
  }
  lds_fence_wave();

  // h_tilde = tanh([ch | h*r] @ Wh + bh); h_new = z*h + (1-z)*h_tilde
#pragma unroll
  for (int nt = 0; nt < 2; ++nt) {
    const int coff = nt * 16;
    floatx8 ah = {};
    ah = wmma_f16(a_ch, loadB(Wh, 0, coff), ah);
    ah = wmma_f16(a_hr, loadB(Wh, 32, coff), ah);
    const float bvh = bh[coff + m];
#pragma unroll
    for (int i = 0; i < 8; ++i) {
      const float ht = tanhf(ah[i] + bvh);
      const float z  = zD[nt][i];
      const float hn = z * hD[nt][i] + (1.0f - z) * ht;
      const int M = i + hi * 8;
      if (active) out_h[(size_t)(base + M) * 32 + coff + m] = hn;
      st[M * 32 + coff + m] = fmaxf(hn, 0.0f);      // relu for head
    }
  }
  lds_fence_wave();

  half16 a_y;
#pragma unroll
  for (int j = 0; j < 8; ++j) {
    a_y[j]     = (_Float16)st[m * 32 + kb8 + j];
    a_y[8 + j] = (_Float16)st[m * 32 + 16 + kb8 + j];
  }

  // y = relu(h_new) @ Wlin + blin
#pragma unroll
  for (int nt = 0; nt < 2; ++nt) {
    const int coff = nt * 16;
    floatx8 ay = {};
    ay = wmma_f16(a_y, loadB(Wlin, 0, coff), ay);
    const float bv = blin[coff + m];
    if (active) {
#pragma unroll
      for (int i = 0; i < 8; ++i) {
        const int M = i + hi * 8;
        out_y[(size_t)(base + M) * 32 + coff + m] = ay[i] + bv;
      }
    }
  }
}

// ---------------------------------------------------------------------------
extern "C" void kernel_launch(void* const* d_in, const int* in_sizes, int n_in,
                              void* d_out, int out_size, void* d_ws, size_t ws_size,
                              hipStream_t stream) {
  const float* X    = (const float*)d_in[0];
  const float* ewt  = (const float*)d_in[1];
  const float* hid  = (const float*)d_in[2];
  const int*   src  = (const int*)d_in[3];
  const int*   dst  = (const int*)d_in[4];
  const float* Wzc  = (const float*)d_in[5];
  const float* bzc  = (const float*)d_in[6];
  const float* Wrc  = (const float*)d_in[7];
  const float* brc  = (const float*)d_in[8];
  const float* Whc  = (const float*)d_in[9];
  const float* bhc  = (const float*)d_in[10];
  const float* Wz   = (const float*)d_in[11];
  const float* bz   = (const float*)d_in[12];
  const float* Wr   = (const float*)d_in[13];
  const float* br   = (const float*)d_in[14];
  const float* Wh   = (const float*)d_in[15];
  const float* bh   = (const float*)d_in[16];
  const float* Wlin = (const float*)d_in[17];
  const float* blin = (const float*)d_in[18];

  float* H3  = (float*)d_ws;                       // [N,96]
  float* AGG = H3 + (size_t)N_NODES * 96;          // [N,96]
  float* out_y = (float*)d_out;                    // [N,32]
  float* out_h = out_y + (size_t)N_NODES * 32;     // [N,32]

  hipMemsetAsync(AGG, 0, (size_t)N_NODES * 96 * sizeof(float), stream);

  const int blocksT = (NT_TILES + WAVES_PER_BLOCK - 1) / WAVES_PER_BLOCK;
  conv_pre_kernel<<<blocksT, 256, 0, stream>>>(X, Wzc, Wrc, Whc, H3);

  const long long totalE = (long long)N_EDGES * 24;
  const int blocksE = (int)((totalE + 255) / 256);
  edge_scatter_kernel<<<blocksE, 256, 0, stream>>>(H3, ewt, src, dst, AGG);

  gates_kernel<<<blocksT, 256, 0, stream>>>(AGG, hid, bzc, brc, bhc,
                                            Wz, bz, Wr, br, Wh, bh, Wlin, blin,
                                            out_y, out_h);
}
